// LayerNormLSTMModel_86655260164133
// MI455X (gfx1250) — matrix operand order repaired and checked by reading, not compiled, not run on hardware
//
#include <hip/hip_runtime.h>
#include <hip/hip_bf16.h>

#define S_LEN 128
#define B_SZ  32
#define E_DIM 512
#define H_DIM 512
#define FH    2048   // 4*H
#define V_SZ  32000
#define L_N   2
#define LN_EPS 1e-5f

typedef __attribute__((ext_vector_type(16))) __bf16 v16bf;
typedef __attribute__((ext_vector_type(8)))  float  v8f;

union FragU { v16bf v; uint4 q[2]; };

__device__ __forceinline__ unsigned short f2bf(float f) {
    unsigned int u = __float_as_uint(f);
    u += 0x7FFFu + ((u >> 16) & 1u);          // round-to-nearest-even
    return (unsigned short)(u >> 16);
}

__device__ __forceinline__ float sigmoidf_(float x) { return 1.0f / (1.0f + expf(-x)); }

__device__ __forceinline__ float block_reduce_sum(float v, float* sred) {
    int t = threadIdx.x;
    sred[t] = v; __syncthreads();
    for (int s = 128; s > 0; s >>= 1) { if (t < s) sred[t] += sred[t + s]; __syncthreads(); }
    float r = sred[0]; __syncthreads();
    return r;
}

// ---------------- fp32 -> bf16 conversion (grid-stride) ----------------
__global__ void cvt_bf16_kernel(const float* __restrict__ in, unsigned short* __restrict__ out, size_t n) {
    size_t i = (size_t)blockIdx.x * blockDim.x + threadIdx.x;
    size_t stride = (size_t)gridDim.x * blockDim.x;
    for (; i < n; i += stride) out[i] = f2bf(in[i]);
}

// ---------------- embedding gather -> bf16 ----------------
__global__ void embed_kernel(const int* __restrict__ tokens, const float* __restrict__ emb,
                             unsigned short* __restrict__ x0) {
    int sb = blockIdx.x;                 // flat [S*B]
    int tok = tokens[sb];
    const float* src = emb + (size_t)tok * E_DIM;
    unsigned short* dst = x0 + (size_t)sb * E_DIM;
    for (int e = threadIdx.x; e < E_DIM; e += blockDim.x) dst[e] = f2bf(src[e]);
}

// ---------------- zero LSTM state ----------------
__global__ void zero_state_kernel(float* __restrict__ c, unsigned short* __restrict__ hbf) {
    int i = blockIdx.x * blockDim.x + threadIdx.x;   // exactly B*H = 16384 threads
    c[i] = 0.0f;
    hbf[i] = 0;
}

// ---------------- per-step dual GEMM: A = x_t@Wi^T, Bm = h@Wh^T -------------
// M=32, N=2048, K=512, bf16 in / f32 out. 512 waves, one 16x16 tile each.
__global__ void lstm_gemm_kernel(const unsigned short* __restrict__ xt,
                                 const unsigned short* __restrict__ hbf,
                                 const unsigned short* __restrict__ Wi,
                                 const unsigned short* __restrict__ Wh,
                                 float* __restrict__ Aout, float* __restrict__ Bout) {
    int wg   = blockIdx.x * 4 + (threadIdx.x >> 5);   // 0..511
    int lane = threadIdx.x & 31;
    int lr = lane & 15, lh = lane >> 4;
    int gemm  = wg >> 8;          // 0: input path, 1: recurrent path
    int rem   = wg & 255;
    int mtile = rem & 1;          // 32 rows -> 2 tiles
    int ntile = rem >> 1;         // 2048 cols -> 128 tiles

    const unsigned short* Asrc = gemm ? hbf : xt;
    const unsigned short* Bsrc = gemm ? Wh  : Wi;
    float* Out = gemm ? Bout : Aout;

    const unsigned short* arow = Asrc + (size_t)(mtile * 16 + lr) * E_DIM;
    const unsigned short* brow = Bsrc + (size_t)(ntile * 16 + lr) * E_DIM;

    v8f acc = {};
    for (int kk = 0; kk < 16; ++kk) {
        int k0 = kk * 32;
        FragU a, b;
        const unsigned short* pa = arow + k0 + 8 * lh;      // A: K = 8*lh+{0..7}, 16+8*lh+{0..7}
        a.q[0] = *(const uint4*)pa;
        a.q[1] = *(const uint4*)(pa + 16);
        const unsigned short* pb = brow + k0 + 8 * lh;      // B: K = 8*lh+{0..7}, 16+8*lh+{0..7}
        b.q[0] = *(const uint4*)pb;
        b.q[1] = *(const uint4*)(pb + 16);
        acc = __builtin_amdgcn_wmma_f32_16x16x32_bf16(false, a.v, false, b.v,
                                                      (short)0, acc, false, false);
    }
    int col = ntile * 16 + lr;
    int rowbase = mtile * 16 + 8 * lh;                       // C layout: M = r + 8*(lane/16)
#pragma unroll
    for (int r = 0; r < 8; ++r)
        Out[(size_t)(rowbase + r) * FH + col] = acc[r];
}

// ---------------- gate math: LN(A)+LN(B), cell update, LN(c), h ------------
__global__ void lstm_gate_kernel(const float* __restrict__ A, const float* __restrict__ Bm,
                                 const float* __restrict__ gi, const float* __restrict__ bi,
                                 const float* __restrict__ gh, const float* __restrict__ bh,
                                 const float* __restrict__ gc, const float* __restrict__ bc,
                                 float* __restrict__ c_state, unsigned short* __restrict__ h_bf,
                                 unsigned short* __restrict__ x_out,
                                 float* __restrict__ hn, float* __restrict__ cn, int last) {
    __shared__ float sg[FH];
    __shared__ float sred[256];
    int b = blockIdx.x, t = threadIdx.x;
    const float* Ar = A  + (size_t)b * FH;
    const float* Br = Bm + (size_t)b * FH;

    float sA = 0.f, sA2 = 0.f, sB = 0.f, sB2 = 0.f;
    for (int n = t; n < FH; n += 256) {
        float a = Ar[n], bb = Br[n];
        sA += a; sA2 += a * a; sB += bb; sB2 += bb * bb;
    }
    sA  = block_reduce_sum(sA,  sred);
    sA2 = block_reduce_sum(sA2, sred);
    sB  = block_reduce_sum(sB,  sred);
    sB2 = block_reduce_sum(sB2, sred);
    const float inv = 1.0f / (float)FH;
    float muA = sA * inv, muB = sB * inv;
    float rsA = rsqrtf(fmaxf(sA2 * inv - muA * muA, 0.f) + LN_EPS);
    float rsB = rsqrtf(fmaxf(sB2 * inv - muB * muB, 0.f) + LN_EPS);

    for (int n = t; n < FH; n += 256)
        sg[n] = (Ar[n] - muA) * rsA * gi[n] + bi[n] + (Br[n] - muB) * rsB * gh[n] + bh[n];
    __syncthreads();

    float cnew[2];
    float sC = 0.f, sC2 = 0.f;
#pragma unroll
    for (int k = 0; k < 2; ++k) {
        int j = t + k * 256;
        float iG = sg[j], fG = sg[H_DIM + j], gG = sg[2 * H_DIM + j];
        float co = c_state[b * H_DIM + j];
        float cv = sigmoidf_(fG) * co + sigmoidf_(iG) * tanhf(gG);
        cnew[k] = cv; sC += cv; sC2 += cv * cv;
    }
    sC  = block_reduce_sum(sC,  sred);
    sC2 = block_reduce_sum(sC2, sred);
    const float invH = 1.0f / (float)H_DIM;
    float muC = sC * invH;
    float rsC = rsqrtf(fmaxf(sC2 * invH - muC * muC, 0.f) + LN_EPS);

#pragma unroll
    for (int k = 0; k < 2; ++k) {
        int j = t + k * 256;
        float cv = cnew[k];
        float hv = sigmoidf_(sg[3 * H_DIM + j]) * tanhf((cv - muC) * rsC * gc[j] + bc[j]);
        c_state[b * H_DIM + j] = cv;
        unsigned short hb = f2bf(hv);
        h_bf[b * H_DIM + j] = hb;
        x_out[(size_t)b * H_DIM + j] = hb;
        if (last) { hn[b * H_DIM + j] = hv; cn[b * H_DIM + j] = cv; }
    }
}

// ---------------- FC GEMM: [4096,512]bf16 x [32000,512]bf16 -> logits fp32 --
// Each wave: a 32x64 slab (2 M-tiles x 4 N-tiles; A and B frags both reused).
// 128 mgroups * 500 ngroups = 64000 waves. Logits stored non-temporal so the
// 524MB stream does not evict the L2-resident 32MB W_fc.
__global__ void fc_gemm_kernel(const unsigned short* __restrict__ X,
                               const unsigned short* __restrict__ W,
                               const float* __restrict__ bias,
                               float* __restrict__ out) {
    int wg   = blockIdx.x * 8 + (threadIdx.x >> 5);  // 0..63999
    int lane = threadIdx.x & 31;
    int lr = lane & 15, lh = lane >> 4;
    int mgroup = wg / 500;                            // 0..127 (32 rows each)
    int ng     = wg - mgroup * 500;                   // 0..499 (64 cols each)

    const unsigned short* xbase = X + (size_t)(mgroup * 32 + lr) * H_DIM;
    v8f acc[2][4] = {};
    for (int kk = 0; kk < 16; ++kk) {
        int k0 = kk * 32;
        FragU a[2];
#pragma unroll
        for (int mt = 0; mt < 2; ++mt) {
            const unsigned short* pa = xbase + (size_t)mt * 16 * H_DIM + k0 + 8 * lh;
            a[mt].q[0] = *(const uint4*)pa;
            a[mt].q[1] = *(const uint4*)(pa + 16);
        }
#pragma unroll
        for (int j = 0; j < 4; ++j) {
            FragU b;
            const unsigned short* pb = W + (size_t)(ng * 64 + j * 16 + lr) * H_DIM + k0 + 8 * lh;
            b.q[0] = *(const uint4*)pb;
            b.q[1] = *(const uint4*)(pb + 16);
#pragma unroll
            for (int mt = 0; mt < 2; ++mt)
                acc[mt][j] = __builtin_amdgcn_wmma_f32_16x16x32_bf16(false, a[mt].v, false, b.v,
                                                                     (short)0, acc[mt][j], false, false);
        }
    }
#pragma unroll
    for (int mt = 0; mt < 2; ++mt) {
        int rowbase = mgroup * 32 + mt * 16 + 8 * lh;
#pragma unroll
        for (int j = 0; j < 4; ++j) {
            int col = ng * 64 + j * 16 + lr;
            float bv = bias[col];
#pragma unroll
            for (int r = 0; r < 8; ++r)
                __builtin_nontemporal_store(acc[mt][j][r] + bv,
                                            out + (size_t)(rowbase + r) * V_SZ + col);
        }
    }
}

// ---------------- in-place log_softmax over V=32000 (online max+sum) -------
__global__ void log_softmax_kernel(float* __restrict__ out) {
    __shared__ float sm[256];
    __shared__ float ss[256];
    int t = threadIdx.x;
    float* p = out + (size_t)blockIdx.x * V_SZ;

    // online pass: running max + scaled sum
    float m = -3.402823466e38f;
    float s = 0.f;
    for (int i = t; i < V_SZ; i += 256) {
        float x = __builtin_nontemporal_load(p + i);
        float mn = fmaxf(m, x);
        s = s * expf(m - mn) + expf(x - mn);
        m = mn;
    }
    sm[t] = m; ss[t] = s; __syncthreads();
    for (int st = 128; st > 0; st >>= 1) {
        if (t < st) {
            float m1 = sm[t], s1 = ss[t];
            float m2 = sm[t + st], s2 = ss[t + st];
            float mn = fmaxf(m1, m2);
            sm[t] = mn;
            ss[t] = s1 * expf(m1 - mn) + s2 * expf(m2 - mn);
        }
        __syncthreads();
    }
    float lse = sm[0] + logf(ss[0]);
    __syncthreads();

    for (int i = t; i < V_SZ; i += 256) {
        float x = __builtin_nontemporal_load(p + i);
        __builtin_nontemporal_store(x - lse, p + i);
    }
}

extern "C" void kernel_launch(void* const* d_in, const int* in_sizes, int n_in,
                              void* d_out, int out_size, void* d_ws, size_t ws_size,
                              hipStream_t stream) {
    const int*   tokens = (const int*)  d_in[0];
    const float* emb    = (const float*)d_in[1];
    const float* W_ih   = (const float*)d_in[2];
    const float* W_hh   = (const float*)d_in[3];
    const float* g_ih   = (const float*)d_in[4];
    const float* b_ih   = (const float*)d_in[5];
    const float* g_hh   = (const float*)d_in[6];
    const float* b_hh   = (const float*)d_in[7];
    const float* g_c    = (const float*)d_in[8];
    const float* b_c    = (const float*)d_in[9];
    const float* W_fc   = (const float*)d_in[10];
    const float* b_fc   = (const float*)d_in[11];
    float* out = (float*)d_out;

    char* ws = (char*)d_ws;
    size_t off = 0;
    auto alloc = [&](size_t bytes) -> void* {
        off = (off + 255) & ~(size_t)255;
        void* p = ws + off;
        off += bytes;
        return p;
    };
    unsigned short* Wi_bf  = (unsigned short*)alloc((size_t)L_N * FH * E_DIM * 2);
    unsigned short* Wh_bf  = (unsigned short*)alloc((size_t)L_N * FH * H_DIM * 2);
    unsigned short* Wfc_bf = (unsigned short*)alloc((size_t)V_SZ * H_DIM * 2);
    unsigned short* x0     = (unsigned short*)alloc((size_t)S_LEN * B_SZ * E_DIM * 2);
    unsigned short* x1     = (unsigned short*)alloc((size_t)S_LEN * B_SZ * H_DIM * 2);
    unsigned short* x2     = (unsigned short*)alloc((size_t)S_LEN * B_SZ * H_DIM * 2);
    float* Amat            = (float*)alloc((size_t)B_SZ * FH * 4);
    float* Bmat            = (float*)alloc((size_t)B_SZ * FH * 4);
    float* c_state         = (float*)alloc((size_t)B_SZ * H_DIM * 4);
    unsigned short* h_bf   = (unsigned short*)alloc((size_t)B_SZ * H_DIM * 2);

    cvt_bf16_kernel<<<2048, 256, 0, stream>>>(W_ih, Wi_bf, (size_t)L_N * FH * E_DIM);
    cvt_bf16_kernel<<<2048, 256, 0, stream>>>(W_hh, Wh_bf, (size_t)L_N * FH * H_DIM);
    cvt_bf16_kernel<<<4096, 256, 0, stream>>>(W_fc, Wfc_bf, (size_t)V_SZ * H_DIM);
    embed_kernel<<<S_LEN * B_SZ, 128, 0, stream>>>(tokens, emb, x0);

    float* hn_base = out + (size_t)S_LEN * B_SZ * V_SZ;
    float* cn_base = hn_base + (size_t)L_N * B_SZ * H_DIM;

    for (int l = 0; l < L_N; ++l) {
        const unsigned short* xin = (l == 0) ? x0 : x1;
        unsigned short* xout      = (l == 0) ? x1 : x2;
        const unsigned short* Wi = Wi_bf + (size_t)l * FH * E_DIM;
        const unsigned short* Wh = Wh_bf + (size_t)l * FH * H_DIM;
        zero_state_kernel<<<64, 256, 0, stream>>>(c_state, h_bf);
        for (int s = 0; s < S_LEN; ++s) {
            lstm_gemm_kernel<<<128, 128, 0, stream>>>(
                xin + (size_t)s * B_SZ * E_DIM, h_bf, Wi, Wh, Amat, Bmat);
            lstm_gate_kernel<<<B_SZ, 256, 0, stream>>>(
                Amat, Bmat,
                g_ih + (size_t)l * FH, b_ih + (size_t)l * FH,
                g_hh + (size_t)l * FH, b_hh + (size_t)l * FH,
                g_c + (size_t)l * H_DIM, b_c + (size_t)l * H_DIM,
                c_state, h_bf, xout + (size_t)s * B_SZ * H_DIM,
                hn_base + (size_t)l * B_SZ * H_DIM,
                cn_base + (size_t)l * B_SZ * H_DIM,
                (s == S_LEN - 1) ? 1 : 0);
        }
    }

    fc_gemm_kernel<<<8000, 256, 0, stream>>>(x2, Wfc_bf, b_fc, out);
    log_softmax_kernel<<<S_LEN * B_SZ, 256, 0, stream>>>(out);
}
